// OptimizedLAM_70205535420697
// MI455X (gfx1250) — compile-verified, matrix operand order checked by reading, not run on hardware
//
#include <hip/hip_runtime.h>
#include <math.h>

// ---------------- model constants ----------------
constexpr int B_  = 8;
constexpr int L_  = 512;
constexpr int D_  = 384;
constexpr int H_  = 12;
constexpr int DH_ = 32;
constexpr int FF_ = 1536;
constexpr int NL_ = 6;
constexpr int BL_ = B_ * L_;   // 4096 tokens

// ---------------- types ----------------
typedef __attribute__((ext_vector_type(16))) __bf16 v16bf;
typedef __attribute__((ext_vector_type(8)))  float  v8f;
typedef __attribute__((ext_vector_type(4)))  unsigned int u32x4;
typedef __attribute__((ext_vector_type(8)))  unsigned int u32x8;

union Frag16 {
    v16bf v;
    unsigned int u[8];
};

// ---------------- small helpers ----------------
__device__ __forceinline__ unsigned short f2bf(float f) {
    unsigned int u = __builtin_bit_cast(unsigned int, f);
    u += 0x7FFFu + ((u >> 16) & 1u);   // round-to-nearest-even
    return (unsigned short)(u >> 16);
}

__device__ __forceinline__ float gelu_exact(float z) {
    return 0.5f * z * (1.f + erff(z * 0.70710678118654752f));
}

__device__ __forceinline__ float sigmoidf_(float z) {
    return 1.f / (1.f + expf(-z));
}

__device__ __forceinline__ float block_reduce_128(float v, float* red) {
    int tid = threadIdx.x;
    __syncthreads();
    red[tid] = v;
    __syncthreads();
    for (int s = 64; s > 0; s >>= 1) {
        if (tid < s) red[tid] += red[tid + s];
        __syncthreads();
    }
    return red[0];
}

// Tensor Data Mover: DMA a [rows x 32] bf16 tile (row stride = strideElems) into LDS.
// D# per CDNA5 ISA: group0 {count,lds_addr,global_addr,type=2},
// group1 {data_size=2B, tensor/tile dims, dim0 stride}. All operands wave-uniform -> SGPRs.
__device__ __forceinline__ void tdm_load_tile(const unsigned short* gptr,
                                              unsigned lds_off,
                                              unsigned rows,
                                              unsigned strideElems) {
    unsigned long long ga = (unsigned long long)(uintptr_t)gptr;
    u32x4 g0;
    g0[0] = 1u;                                              // count=1, user mode
    g0[1] = lds_off;                                         // lds_addr (bytes)
    g0[2] = (unsigned)ga;                                    // global_addr[31:0]
    g0[3] = (unsigned)((ga >> 32) & 0x01FFFFFFu) | (2u << 30); // addr[56:32] | type=2
    unsigned long long st = (unsigned long long)strideElems;
    u32x8 g1;
    g1[0] = (1u << 16);                                      // data_size=1 (2 bytes), mask=0
    g1[1] = (32u & 0xFFFFu) << 16;                           // tensor_dim0[15:0]=32
    g1[2] = 0u | ((rows & 0xFFFFu) << 16);                   // dim0 hi | tensor_dim1 lo
    g1[3] = ((rows >> 16) & 0xFFFFu) | (32u << 16);          // dim1 hi | tile_dim0=32
    g1[4] = rows & 0xFFFFu;                                  // tile_dim1=rows, tile_dim2=0
    g1[5] = (unsigned)(st & 0xFFFFFFFFu);                    // dim0_stride lo
    g1[6] = (unsigned)((st >> 32) & 0xFFFFu);                // dim0_stride hi
    g1[7] = 0u;
    asm volatile("tensor_load_to_lds %0, %1" :: "s"(g0), "s"(g1) : "memory");
}

// ---------------- f32 -> bf16 transposed weight staging: dst[N,K] = src[K,N] ----------------
__global__ void f2bf_t_kernel(const float* __restrict__ src,
                              unsigned short* __restrict__ dst, int Kd, int Nd) {
    int i = blockIdx.x * blockDim.x + threadIdx.x;
    if (i < Kd * Nd) {
        int k = i / Nd, n = i % Nd;
        dst[(size_t)n * Kd + k] = f2bf(src[i]);
    }
}

// ---------------- embedding + LayerNorm ----------------
__global__ __launch_bounds__(128)
void embed_ln_kernel(const int* __restrict__ ids,
                     const float* __restrict__ we, const float* __restrict__ pe,
                     const float* __restrict__ te,
                     const float* __restrict__ lw, const float* __restrict__ lb,
                     float* __restrict__ x, unsigned short* __restrict__ xb) {
    __shared__ float red[128];
    int tok = blockIdx.x;
    int tid = threadIdx.x;
    int l   = tok % L_;
    int id  = ids[tok];
    float v0[3];
    float s = 0.f, sq = 0.f;
#pragma unroll
    for (int r = 0; r < 3; ++r) {
        int d = tid + r * 128;
        float t = we[(size_t)id * D_ + d] + pe[(size_t)l * D_ + d] + te[d];
        v0[r] = t;
        s += t; sq += t * t;
    }
    s  = block_reduce_128(s, red);
    sq = block_reduce_128(sq, red);
    float mu  = s * (1.f / D_);
    float var = fmaxf(sq * (1.f / D_) - mu * mu, 0.f);
    float inv = rsqrtf(var + 1e-12f);
#pragma unroll
    for (int r = 0; r < 3; ++r) {
        int d = tid + r * 128;
        float y = (v0[r] - mu) * inv * lw[d] + lb[d];
        x[(size_t)tok * D_ + d]  = y;
        xb[(size_t)tok * D_ + d] = f2bf(y);
    }
}

// ---------------- residual + LayerNorm (in-place on x, refresh bf16 shadow) ----------------
__global__ __launch_bounds__(128)
void ln_residual_kernel(float* __restrict__ x, const float* __restrict__ delta,
                        const float* __restrict__ lw, const float* __restrict__ lb,
                        unsigned short* __restrict__ xb) {
    __shared__ float red[128];
    int tok = blockIdx.x;
    int tid = threadIdx.x;
    float v0[3];
    float s = 0.f, sq = 0.f;
#pragma unroll
    for (int r = 0; r < 3; ++r) {
        int d = tid + r * 128;
        float t = x[(size_t)tok * D_ + d] + delta[(size_t)tok * D_ + d];
        v0[r] = t;
        s += t; sq += t * t;
    }
    s  = block_reduce_128(s, red);
    sq = block_reduce_128(sq, red);
    float mu  = s * (1.f / D_);
    float var = fmaxf(sq * (1.f / D_) - mu * mu, 0.f);
    float inv = rsqrtf(var + 1e-12f);
#pragma unroll
    for (int r = 0; r < 3; ++r) {
        int d = tid + r * 128;
        float y = (v0[r] - mu) * inv * lw[d] + lb[d];
        x[(size_t)tok * D_ + d]  = y;
        xb[(size_t)tok * D_ + d] = f2bf(y);
    }
}

// ---------------- TDM-staged WMMA GEMM ----------------
// C[M,N] = act(A[M,K] @ W[K,N] + bias), weights given transposed WT[N,K] (bf16).
// Block = 256 threads = 8 waves. Block tile 128x64; wave w owns rows w*16..w*16+15 and
// all 4 column tiles (1 A-frag reused over 4 B-frags -> 4 WMMAs per K-step per wave).
// A & B panels are DMA'd into LDS by the Tensor Data Mover (double buffered).
template <int ACT, int WF32, int WBF16>
__global__ __launch_bounds__(256)
void gemm_tdm_kernel(const unsigned short* __restrict__ A,
                     const unsigned short* __restrict__ WT,
                     const float* __restrict__ bias,
                     float* __restrict__ Cf,
                     unsigned short* __restrict__ Cb,
                     int M, int N, int K) {
    __shared__ unsigned int ldsA[2][128 * 16];  // 128 rows x 32 bf16 (64B rows) x2 buffers
    __shared__ unsigned int ldsB[2][64 * 16];   // 64 rows  x 32 bf16 x2 buffers

    int tid  = threadIdx.x;
    int lane = tid & 31;
    int w    = tid >> 5;
    int m0   = blockIdx.x * 128;
    int n0   = blockIdx.y * 64;

    v8f acc[4];
#pragma unroll
    for (int nt = 0; nt < 4; ++nt) acc[nt] = v8f{0.f, 0.f, 0.f, 0.f, 0.f, 0.f, 0.f, 0.f};

    const unsigned short* Abase  = A  + (size_t)m0 * K;
    const unsigned short* WTbase = WT + (size_t)n0 * K;

    // prologue: DMA first K-chunk (TDM, tracked by TENSORcnt)
    if (tid == 0) {
        tdm_load_tile(Abase,  (unsigned)(uintptr_t)&ldsA[0][0], 128u, (unsigned)K);
        tdm_load_tile(WTbase, (unsigned)(uintptr_t)&ldsB[0][0], 64u,  (unsigned)K);
    }

    int cur = 0;
    int kh2 = (lane >> 4) * 4;          // dword sub-offset for lane half
    int rA  = w * 16 + (lane & 15);     // this wave's A row in the 128-row panel

    for (int kb = 0; kb < K; kb += 32) {
        if (tid == 0) __builtin_amdgcn_s_wait_tensorcnt(0);
        __syncthreads();
        if (tid == 0 && kb + 32 < K) {
            tdm_load_tile(Abase  + kb + 32, (unsigned)(uintptr_t)&ldsA[cur ^ 1][0], 128u, (unsigned)K);
            tdm_load_tile(WTbase + kb + 32, (unsigned)(uintptr_t)&ldsB[cur ^ 1][0], 64u,  (unsigned)K);
        }

        Frag16 af;
        const unsigned int* arow = &ldsA[cur][rA * 16];
        *(uint4*)&af.u[0] = *(const uint4*)(arow + kh2);
        *(uint4*)&af.u[4] = *(const uint4*)(arow + 8 + kh2);

#pragma unroll
        for (int nt = 0; nt < 4; ++nt) {
            Frag16 bf;
            const unsigned int* brow = &ldsB[cur][(nt * 16 + (lane & 15)) * 16];
            *(uint4*)&bf.u[0] = *(const uint4*)(brow + kh2);
            *(uint4*)&bf.u[4] = *(const uint4*)(brow + 8 + kh2);
            acc[nt] = __builtin_amdgcn_wmma_f32_16x16x32_bf16(
                false, af.v, false, bf.v, (short)0, acc[nt], false, false);
        }
        __syncthreads();
        cur ^= 1;
    }

    // epilogue: bias + activation + store (C/D layout per ISA: VGPR r -> row r / r+8)
    int rbase = m0 + w * 16 + ((lane >> 4) * 8);
#pragma unroll
    for (int nt = 0; nt < 4; ++nt) {
        int ncol = n0 + nt * 16 + (lane & 15);
        float bv = bias ? bias[ncol] : 0.f;
#pragma unroll
        for (int r = 0; r < 8; ++r) {
            float s = acc[nt][r] + bv;
            if (ACT == 1) s = s * sigmoidf_(s);           // silu
            if (ACT == 2) s = gelu_exact(gelu_exact(s));  // gelu(gelu(.))
            size_t idx = (size_t)(rbase + r) * N + ncol;
            if (WF32)  Cf[idx] = s;
            if (WBF16) Cb[idx] = f2bf(s);
        }
    }
}

// ---------------- small gate projections: sigmoid(x @ W[D,12] + b (+extra)) ----------------
__global__ __launch_bounds__(256)
void gateproj_kernel(const float* __restrict__ x, const float* __restrict__ W,
                     const float* __restrict__ bvec, const float* __restrict__ extra,
                     float* __restrict__ out) {
    int idx = blockIdx.x * blockDim.x + threadIdx.x;
    if (idx >= BL_ * H_) return;
    int tok = idx / H_, n = idx % H_;
    const float* xr = x + (size_t)tok * D_;
    float s = bvec[n] + (extra ? extra[n] : 0.f);
#pragma unroll 4
    for (int d = 0; d < D_; ++d) s += xr[d] * W[d * H_ + n];
    out[idx] = sigmoidf_(s);
}

// ---------------- delta-rule scan: one wave32 per (b,h), lane e owns column e ----------------
__global__ __launch_bounds__(32)
void scan_kernel(const float* __restrict__ qf, const float* __restrict__ kf,
                 const float* __restrict__ vf,
                 const float* __restrict__ beta, const float* __restrict__ fd,
                 const float* __restrict__ sd, const float* __restrict__ fg,
                 const float* __restrict__ sg, const float* __restrict__ flux,
                 float* __restrict__ obuf) {
    int bh = blockIdx.x;
    int b = bh / H_, h = bh % H_;
    int e = threadIdx.x;
    float Sf[DH_], Ss[DH_];
#pragma unroll
    for (int d = 0; d < DH_; ++d) { Sf[d] = 0.f; Ss[d] = 0.f; }
    __shared__ float shq[DH_], shk[DH_];
    float fx = flux[h];

    for (int t = 0; t < L_; ++t) {
        int tok  = b * L_ + t;
        int base = tok * D_ + h * DH_;
        float qe = qf[base + e], ke = kf[base + e], ve = vf[base + e];
        float qs = qe * qe, ks = ke * ke;
#pragma unroll
        for (int m = 16; m >= 1; m >>= 1) {
            qs += __shfl_xor(qs, m, 32);
            ks += __shfl_xor(ks, m, 32);
        }
        qe *= rsqrtf(qs + 1e-8f);
        ke *= rsqrtf(ks + 1e-8f);
        shq[e] = qe; shk[e] = ke;
        __syncthreads();

        int gi = tok * H_ + h;
        float bt = beta[gi], fdt = fd[gi], sdt = sd[gi], fgt = fg[gi], sgt = sg[gi];

        float dotf = 0.f, dots = 0.f;
#pragma unroll
        for (int d = 0; d < DH_; ++d) {
            Sf[d] *= fdt; dotf += shk[d] * Sf[d];
            Ss[d] *= sdt; dots += shk[d] * Ss[d];
        }
        float bef = bt * (ve - dotf);
        float bes = bt * (ve - dots);
        float of = 0.f, os = 0.f;
#pragma unroll
        for (int d = 0; d < DH_; ++d) {
            Sf[d] += shk[d] * bef; of += shq[d] * Sf[d];
            Ss[d] += shk[d] * bes; os += shq[d] * Ss[d];
        }
        obuf[base + e] = fgt * of + sgt * os + fx * of * os;
        __syncthreads();
    }
}

// ---------------- per-head RMSNorm + swish gate -> bf16 ----------------
__global__ __launch_bounds__(256)
void gate_out_kernel(const float* __restrict__ obuf, const float* __restrict__ gbuf,
                     const float* __restrict__ onw, unsigned short* __restrict__ obf) {
    int idx = blockIdx.x * 256 + threadIdx.x;
    int grp = idx >> 5, e = idx & 31;
    if (grp >= BL_ * H_) return;
    int tok = grp / H_, h = grp % H_;
    size_t pos = (size_t)tok * D_ + h * DH_ + e;
    float o  = obuf[pos];
    float ms = o * o;
#pragma unroll
    for (int m = 16; m >= 1; m >>= 1) ms += __shfl_xor(ms, m, 32);
    ms *= (1.f / DH_);
    o *= rsqrtf(ms + 1e-6f) * onw[e];
    float g = gbuf[pos];
    o *= g * sigmoidf_(g);
    obf[pos] = f2bf(o);
}

// ---------------- masked mean-pool + L2 normalize ----------------
__global__ __launch_bounds__(384)
void pool_kernel(const float* __restrict__ x, const int* __restrict__ mask,
                 float* __restrict__ out) {
    __shared__ float sh[D_];
    int b = blockIdx.x, d = threadIdx.x;
    float s = 0.f, cnt = 0.f;
    for (int l = 0; l < L_; ++l) {
        float m = (float)mask[b * L_ + l];
        s   += x[((size_t)(b * L_ + l)) * D_ + d] * m;
        cnt += m;
    }
    cnt = fmaxf(cnt, 1e-9f);
    float e = s / cnt;
    sh[d] = e * e;
    __syncthreads();
    if (d < 128) sh[d] += sh[d + 128] + sh[d + 256];
    __syncthreads();
    for (int s2 = 64; s2 > 0; s2 >>= 1) {
        if (d < s2) sh[d] += sh[d + s2];
        __syncthreads();
    }
    float nrm = sqrtf(sh[0]);
    out[(size_t)b * D_ + d] = e / fmaxf(nrm, 1e-12f);
}

// =====================================================================
extern "C" void kernel_launch(void* const* d_in, const int* in_sizes, int n_in,
                              void* d_out, int out_size, void* d_ws, size_t ws_size,
                              hipStream_t stream) {
    const float* word_emb = (const float*)d_in[0];
    const float* pos_emb  = (const float*)d_in[1];
    const float* type_emb = (const float*)d_in[2];
    const float* emb_ln_w = (const float*)d_in[3];
    const float* emb_ln_b = (const float*)d_in[4];
    const float* qW = (const float*)d_in[5];
    const float* kW = (const float*)d_in[6];
    const float* vW = (const float*)d_in[7];
    const float* gW = (const float*)d_in[8];
    const float* oW = (const float*)d_in[9];
    const float* qb = (const float*)d_in[10];
    const float* kb = (const float*)d_in[11];
    const float* vb = (const float*)d_in[12];
    const float* gb = (const float*)d_in[13];
    const float* ob = (const float*)d_in[14];
    const float* bW  = (const float*)d_in[15];
    const float* fdW = (const float*)d_in[16];
    const float* sdW = (const float*)d_in[17];
    const float* fgW = (const float*)d_in[18];
    const float* sgW = (const float*)d_in[19];
    const float* bb  = (const float*)d_in[20];
    const float* fdb = (const float*)d_in[21];
    const float* sdb = (const float*)d_in[22];
    const float* fgb = (const float*)d_in[23];
    const float* sgb = (const float*)d_in[24];
    const float* fd_bias = (const float*)d_in[25];
    const float* sd_bias = (const float*)d_in[26];
    const float* flux    = (const float*)d_in[27];
    const float* onorm_w = (const float*)d_in[28];
    const float* ln1_w = (const float*)d_in[29];
    const float* ln2_w = (const float*)d_in[30];
    const float* ln1_b = (const float*)d_in[31];
    const float* ln2_b = (const float*)d_in[32];
    const float* interW = (const float*)d_in[33];
    const float* interb = (const float*)d_in[34];
    const float* outW   = (const float*)d_in[35];
    const float* outb   = (const float*)d_in[36];
    const int* input_ids      = (const int*)d_in[37];
    const int* attention_mask = (const int*)d_in[38];
    float* out = (float*)d_out;

    // -------- workspace carve --------
    char* base = (char*)d_ws;
    size_t off = 0;
    auto alloc = [&](size_t bytes) -> void* {
        void* p = base + off;
        off += (bytes + 255) & ~(size_t)255;
        return p;
    };
    float* x   = (float*)alloc((size_t)BL_ * D_ * 4);
    unsigned short* xb = (unsigned short*)alloc((size_t)BL_ * D_ * 2);
    float* qf  = (float*)alloc((size_t)BL_ * D_ * 4);
    float* kf  = (float*)alloc((size_t)BL_ * D_ * 4);
    float* vf  = (float*)alloc((size_t)BL_ * D_ * 4);
    float* gf  = (float*)alloc((size_t)BL_ * D_ * 4);
    float* gbeta = (float*)alloc((size_t)BL_ * H_ * 4);
    float* gfd   = (float*)alloc((size_t)BL_ * H_ * 4);
    float* gsd   = (float*)alloc((size_t)BL_ * H_ * 4);
    float* gfg   = (float*)alloc((size_t)BL_ * H_ * 4);
    float* gsg   = (float*)alloc((size_t)BL_ * H_ * 4);
    float* obuf  = (float*)alloc((size_t)BL_ * D_ * 4);
    unsigned short* obf = (unsigned short*)alloc((size_t)BL_ * D_ * 2);
    float* tmp   = (float*)alloc((size_t)BL_ * D_ * 4);
    unsigned short* hb = (unsigned short*)alloc((size_t)BL_ * FF_ * 2);
    // transposed bf16 weights for this layer: WT[N,K]
    unsigned short* wq = (unsigned short*)alloc((size_t)D_ * D_ * 2);
    unsigned short* wk = (unsigned short*)alloc((size_t)D_ * D_ * 2);
    unsigned short* wv = (unsigned short*)alloc((size_t)D_ * D_ * 2);
    unsigned short* wg = (unsigned short*)alloc((size_t)D_ * D_ * 2);
    unsigned short* wo = (unsigned short*)alloc((size_t)D_ * D_ * 2);
    unsigned short* wi = (unsigned short*)alloc((size_t)D_ * FF_ * 2);  // [FF, D]
    unsigned short* wu = (unsigned short*)alloc((size_t)FF_ * D_ * 2);  // [D, FF]

    embed_ln_kernel<<<BL_, 128, 0, stream>>>(input_ids, word_emb, pos_emb, type_emb,
                                             emb_ln_w, emb_ln_b, x, xb);

    const int DD = D_ * D_;
    const int DF = D_ * FF_;
    dim3 gD(BL_ / 128, D_ / 64);     // 32 x 6
    dim3 gF(BL_ / 128, FF_ / 64);    // 32 x 24

    for (int i = 0; i < NL_; ++i) {
        // stage this layer's weights as transposed bf16 (WT[N,K])
        f2bf_t_kernel<<<(DD + 255) / 256, 256, 0, stream>>>(qW + (size_t)i * DD, wq, D_, D_);
        f2bf_t_kernel<<<(DD + 255) / 256, 256, 0, stream>>>(kW + (size_t)i * DD, wk, D_, D_);
        f2bf_t_kernel<<<(DD + 255) / 256, 256, 0, stream>>>(vW + (size_t)i * DD, wv, D_, D_);
        f2bf_t_kernel<<<(DD + 255) / 256, 256, 0, stream>>>(gW + (size_t)i * DD, wg, D_, D_);
        f2bf_t_kernel<<<(DD + 255) / 256, 256, 0, stream>>>(oW + (size_t)i * DD, wo, D_, D_);
        f2bf_t_kernel<<<(DF + 255) / 256, 256, 0, stream>>>(interW + (size_t)i * DF, wi, D_, FF_);
        f2bf_t_kernel<<<(DF + 255) / 256, 256, 0, stream>>>(outW + (size_t)i * DF, wu, FF_, D_);

        // q, k, v with fused silu; g plain
        gemm_tdm_kernel<1, 1, 0><<<gD, 256, 0, stream>>>(xb, wq, qb + i * D_, qf, nullptr, BL_, D_, D_);
        gemm_tdm_kernel<1, 1, 0><<<gD, 256, 0, stream>>>(xb, wk, kb + i * D_, kf, nullptr, BL_, D_, D_);
        gemm_tdm_kernel<1, 1, 0><<<gD, 256, 0, stream>>>(xb, wv, vb + i * D_, vf, nullptr, BL_, D_, D_);
        gemm_tdm_kernel<0, 1, 0><<<gD, 256, 0, stream>>>(xb, wg, gb + i * D_, gf, nullptr, BL_, D_, D_);

        // gate projections (sigmoid, optional decay bias)
        int gblocks = (BL_ * H_ + 255) / 256;
        gateproj_kernel<<<gblocks, 256, 0, stream>>>(x, bW  + (size_t)i * D_ * H_, bb  + i * H_, nullptr,          gbeta);
        gateproj_kernel<<<gblocks, 256, 0, stream>>>(x, fdW + (size_t)i * D_ * H_, fdb + i * H_, fd_bias + i * H_, gfd);
        gateproj_kernel<<<gblocks, 256, 0, stream>>>(x, sdW + (size_t)i * D_ * H_, sdb + i * H_, sd_bias + i * H_, gsd);
        gateproj_kernel<<<gblocks, 256, 0, stream>>>(x, fgW + (size_t)i * D_ * H_, fgb + i * H_, nullptr,          gfg);
        gateproj_kernel<<<gblocks, 256, 0, stream>>>(x, sgW + (size_t)i * D_ * H_, sgb + i * H_, nullptr,          gsg);

        // recurrent delta-rule scan: 96 waves, register-resident dual state
        scan_kernel<<<B_ * H_, 32, 0, stream>>>(qf, kf, vf, gbeta, gfd, gsd, gfg, gsg,
                                                flux + i * H_, obuf);

        // per-head RMSNorm + swish gate -> bf16 operand for output projection
        gate_out_kernel<<<(BL_ * H_ * 32) / 256, 256, 0, stream>>>(obuf, gf, onorm_w + i * DH_, obf);

        // output projection, residual + LN1
        gemm_tdm_kernel<0, 1, 0><<<gD, 256, 0, stream>>>(obf, wo, ob + i * D_, tmp, nullptr, BL_, D_, D_);
        ln_residual_kernel<<<BL_, 128, 0, stream>>>(x, tmp, ln1_w + i * D_, ln1_b + i * D_, xb);

        // FFN: double-gelu intermediate (bf16 only), then output, residual + LN2
        gemm_tdm_kernel<2, 0, 1><<<gF, 256, 0, stream>>>(xb, wi, interb + i * FF_, nullptr, hb, BL_, FF_, D_);
        gemm_tdm_kernel<0, 1, 0><<<gD, 256, 0, stream>>>(hb, wu, outb + i * D_, tmp, nullptr, BL_, D_, FF_);
        ln_residual_kernel<<<BL_, 128, 0, stream>>>(x, tmp, ln2_w + i * D_, ln2_b + i * D_, xb);
    }

    pool_kernel<<<B_, D_, 0, stream>>>(x, attention_mask, out);

    (void)in_sizes; (void)n_in; (void)out_size; (void)ws_size;
}
